// SageLayer_841813590040
// MI455X (gfx1250) — matrix operand order repaired and checked by reading, not compiled, not run on hardware
//
#include <hip/hip_runtime.h>
#include <hip/hip_bf16.h>

typedef __attribute__((ext_vector_type(16))) _Float16 v16h;
typedef __attribute__((ext_vector_type(8)))  _Float16 v8h;
typedef __attribute__((ext_vector_type(4)))  _Float16 v4h;
typedef __attribute__((ext_vector_type(8)))  float    v8f;
typedef __attribute__((ext_vector_type(4)))  float    v4f;

#define D_IN      128
#define D3        128
#define D_OUT     384
#define K_NEIGH   32
#define NEG_SLOPE 0.2f
#define EPS       1e-12f

// ---------------------------------------------------------------------------
// Pre-pack the three 128x128 f32 weight matrices into f16 WMMA B-fragments.
// Fragment layout (dense 16-bit B 32x16, contiguous-K order chosen here, same as A): lane L holds
// column N = L%16, K-halfblock = (L/16)*16, 16 contiguous K halves per lane.
// dst index: [branch][ktile(4)][ntile(8)][lane(32)][16 halves], contiguous.
// ---------------------------------------------------------------------------
__global__ __launch_bounds__(32) void pack_weights(
    const float* __restrict__ W0, const float* __restrict__ W1,
    const float* __restrict__ W2, _Float16* __restrict__ dst)
{
    int frag = blockIdx.x;          // 0..95 = branch*32 + ktile*8 + ntile
    int br   = frag >> 5;
    int rem  = frag & 31;
    int kt   = rem >> 3;
    int nt   = rem & 7;
    const float* W = (br == 0) ? W0 : (br == 1) ? W1 : W2;

    int lane = threadIdx.x;
    int n  = nt * 16 + (lane & 15);
    int kb = kt * 32 + ((lane >> 4) << 4);   // K base for this lane

    _Float16* p = dst + ((size_t)frag * 32 + lane) * 16;
#pragma unroll
    for (int h = 0; h < 16; ++h)
        p[h] = (_Float16)W[(size_t)(kb + h) * D3 + n];
}

// ---------------------------------------------------------------------------
// Fused: gather+mean aggregation (b128 row reads) -> f16 LDS -> WMMA GEMM vs
// packed weights -> bias + LeakyReLU -> row L2 normalize -> store.
// One block = 16 nodes = one WMMA M-tile. 128 threads = 4 waves.
// ---------------------------------------------------------------------------
__global__ __launch_bounds__(128) void sage_fused(
    const int*      __restrict__ nodes,
    const int*      __restrict__ adj,
    const int*      __restrict__ dis,
    const float*    __restrict__ table,
    const _Float16* __restrict__ packedW,
    const float*    __restrict__ bias,
    float*          __restrict__ out,
    int n_nodes)
{
    __shared__ _Float16 hA[3][16][D_IN];        // A matrices (f16), 12 KB
    __shared__ int      sIdx[2][16][K_NEIGH];   // neighbor ids, 4 KB
    __shared__ int      sNodes[16];
    __shared__ float    feats[16][D_OUT + 4];   // padded vs LDS bank conflicts
    __shared__ float    partial[16][8];
    __shared__ float    invnorm[16];

    const int t    = threadIdx.x;               // 0..127
    const int wave = t >> 5;
    const int lane = t & 31;
    const int base = blockIdx.x * 16;

    // ---- stage indices into LDS (coalesced) ----
    if (t < 16) {
        int gi = base + t;
        sNodes[t] = nodes[gi < n_nodes ? gi : n_nodes - 1];
    }
    for (int j = t; j < 16 * K_NEIGH; j += 128) {
        int gi = base * K_NEIGH + j;
        int gc = gi < n_nodes * K_NEIGH ? gi : n_nodes * K_NEIGH - 1;
        sIdx[0][j / K_NEIGH][j % K_NEIGH] = adj[gc];
        sIdx[1][j / K_NEIGH][j % K_NEIGH] = dis[gc];
    }
    __syncthreads();

    // ---- gather + mean aggregation ----
    // Each wave owns 4 nodes; each gathered 512 B table row is read by ONE
    // global_load_b128 per lane (32 lanes x 16 B = full row, fully coalesced).
    // Table (51.2 MB) is L2-resident (192 MB L2), so this is an L2-BW phase;
    // b128 reads quarter the VMEM instruction count vs scalar loads.
    {
        const int c4 = lane * 4;                // this lane's 4 columns
        for (int ii = 0; ii < 4; ++ii) {
            const int i = wave * 4 + ii;
            v4f s = *(const v4f*)&table[(size_t)sNodes[i] * D_IN + c4];
            v4f a = (v4f)0.f, d = (v4f)0.f;
#pragma unroll 8
            for (int k = 0; k < K_NEIGH; ++k)
                a += *(const v4f*)&table[(size_t)sIdx[0][i][k] * D_IN + c4];
#pragma unroll 8
            for (int k = 0; k < K_NEIGH; ++k)
                d += *(const v4f*)&table[(size_t)sIdx[1][i][k] * D_IN + c4];

            v4h hs, ha, hd;
#pragma unroll
            for (int j = 0; j < 4; ++j) {
                hs[j] = (_Float16)s[j];
                ha[j] = (_Float16)(a[j] * (1.f / 32.f));
                hd[j] = (_Float16)(d[j] * (1.f / 32.f));
            }
            *(v4h*)&hA[0][i][c4] = hs;          // one ds_store_b64 each
            *(v4h*)&hA[1][i][c4] = ha;
            *(v4h*)&hA[2][i][c4] = hd;
        }
    }
    __syncthreads();

    // ---- WMMA GEMM: 24 output tiles (3 branches x 8 ntiles), 6 per wave ----
    union AFrag { v16h v; v8h h[2]; };

    for (int tile = wave; tile < 24; tile += 4) {
        const int br = tile >> 3;
        const int nt = tile & 7;
        v8f acc = {};
#pragma unroll
        for (int kt = 0; kt < 4; ++kt) {
            // A fragment: lane%16 = M,
            // low lanes hold K=0..15, high lanes K=16..31 (matches packed B).
            const int m  = lane & 15;
            const int kb = kt * 32 + ((lane >> 4) << 4);
            AFrag af;
            af.h[0] = *(const v8h*)&hA[br][m][kb];
            af.h[1] = *(const v8h*)&hA[br][m][kb + 8];

            // B fragment: contiguous 32 B per lane from the pre-packed buffer.
            const v16h bf = *(const v16h*)(packedW +
                ((size_t)(br * 32 + kt * 8 + nt) * 32 + lane) * 16);

            acc = __builtin_amdgcn_wmma_f32_16x16x32_f16(
                false, af.v, false, bf, (short)0, acc, false, false);
        }
        // C/D layout: VGPR r -> M = r + 8*(lane/16), N = lane%16.
#pragma unroll
        for (int r = 0; r < 8; ++r) {
            const int m = r + ((lane >> 4) << 3);
            const int n = br * D3 + nt * 16 + (lane & 15);
            float v = acc[r] + bias[n];
            v = (v >= 0.f) ? v : NEG_SLOPE * v;
            feats[m][n] = v;
        }
    }
    __syncthreads();

    // ---- row L2 norm: 8 partial lanes per row, then combine ----
    {
        const int row = t >> 3, part = t & 7;
        float s = 0.f;
        for (int j = part; j < D_OUT; j += 8) {
            const float v = feats[row][j];
            s += v * v;
        }
        partial[row][part] = s;
    }
    __syncthreads();
    if (t < 16) {
        float s = 0.f;
#pragma unroll
        for (int p = 0; p < 8; ++p) s += partial[t][p];
        invnorm[t] = 1.f / fmaxf(sqrtf(s), EPS);
    }
    __syncthreads();

    // ---- coalesced store: 48 elements per thread ----
    for (int j = t; j < 16 * D_OUT; j += 128) {
        const int i = j / D_OUT, col = j % D_OUT;
        if (base + i < n_nodes)
            out[(size_t)(base + i) * D_OUT + col] = feats[i][col] * invnorm[i];
    }
}

extern "C" void kernel_launch(void* const* d_in, const int* in_sizes, int n_in,
                              void* d_out, int out_size, void* d_ws, size_t ws_size,
                              hipStream_t stream) {
    const int*   nodes = (const int*)  d_in[0];
    const int*   adj   = (const int*)  d_in[1];
    const int*   dis   = (const int*)  d_in[2];
    const float* table = (const float*)d_in[3];
    const float* Wself = (const float*)d_in[4];
    const float* Wadj  = (const float*)d_in[5];
    const float* Wdis  = (const float*)d_in[6];
    const float* bias  = (const float*)d_in[7];

    const int n_nodes = in_sizes[0];

    // 3 branches * 32 fragments * 32 lanes * 16 halves = 49152 f16 = 96 KB
    _Float16* packedW = (_Float16*)d_ws;
    pack_weights<<<96, 32, 0, stream>>>(Wself, Wadj, Wdis, packedW);

    const int nblocks = (n_nodes + 15) / 16;   // 3125 for N=50000
    sage_fused<<<nblocks, 128, 0, stream>>>(
        nodes, adj, dis, table, packedW, bias, (float*)d_out, n_nodes);
}
